// CausalSelfAttention_10711648436898
// MI455X (gfx1250) — matrix operand order repaired and checked by reading, not compiled, not run on hardware
//
#include <hip/hip_runtime.h>
#include <hip/hip_bf16.h>
#include <math.h>

typedef __attribute__((ext_vector_type(16))) __bf16 v16bf;
typedef __attribute__((ext_vector_type(8)))  __bf16 v8bf;
typedef __attribute__((ext_vector_type(4)))  __bf16 v4bf;
typedef __attribute__((ext_vector_type(8)))  float  v8f;
typedef __attribute__((ext_vector_type(4)))  unsigned int u32x4;
typedef __attribute__((ext_vector_type(8)))  int i32x8;
typedef __attribute__((ext_vector_type(4)))  int i32x4;

#define D_MODEL  1024
#define N_HEADS  16
#define HEAD_DIM 64
#define TSEQ     2048
#define BATCH    2
#define MTOT     (BATCH * TSEQ)   // 4096 rows total

#if defined(__has_builtin)
#if __has_builtin(__builtin_amdgcn_tensor_load_to_lds) && __has_builtin(__builtin_amdgcn_s_wait_tensorcnt)
#define HAVE_TDM 1
#endif
#endif
#ifndef HAVE_TDM
#define HAVE_TDM 0
#endif

union FragU { v16bf v; v8bf h[2]; };

__device__ __forceinline__ v16bf make_frag(v8bf lo, v8bf hi) {
    FragU u; u.h[0] = lo; u.h[1] = hi; return u.v;
}

#if HAVE_TDM
// Issue a 2D TDM tile load: tile_y rows of tile_x elements (2-byte data),
// row stride stride_x elements, into LDS at byte offset lds_off.
// D# layout per CDNA5 ISA 8.3/8.4 (group0: count/lds/global/type, group1: dims).
__device__ __forceinline__ void tdm_load_2d(unsigned int lds_off, const void* gaddr,
                                            unsigned int tile_x, unsigned int tile_y,
                                            unsigned int tensor_x, unsigned int tensor_y,
                                            unsigned int stride_x) {
    unsigned long long ga = (unsigned long long)(size_t)gaddr;
    u32x4 g0;
    g0[0] = 1u;                                   // count=1 (valid), user mode
    g0[1] = lds_off;                              // lds_addr (bytes)
    g0[2] = (unsigned int)ga;                     // global_addr[31:0]
    g0[3] = (unsigned int)(ga >> 32) | (2u << 30);// global_addr[56:32] | type=2
    i32x8 g1;
    g1[0] = (int)(1u << 16);                      // wg_mask=0, data_size=1 (2B)
    g1[1] = (int)((tensor_x & 0xffffu) << 16);    // tensor_dim0[15:0]
    g1[2] = (int)((tensor_x >> 16) | ((tensor_y & 0xffffu) << 16)); // dim0 hi | dim1 lo
    g1[3] = (int)((tensor_y >> 16) | (tile_x << 16));               // dim1 hi | tile_dim0
    g1[4] = (int)tile_y;                          // tile_dim1 (tile_dim2=0)
    g1[5] = (int)stride_x;                        // tensor_dim0_stride[31:0]
    g1[6] = 0;                                    // stride0 hi | dim1_stride lo
    g1[7] = 0;
    i32x4 z4 = {0, 0, 0, 0};
#if __clang_major__ >= 23
    i32x8 z8 = {0, 0, 0, 0, 0, 0, 0, 0};
    __builtin_amdgcn_tensor_load_to_lds(g0, g1, z4, z4, z8, 0);
#else
    __builtin_amdgcn_tensor_load_to_lds(g0, g1, z4, z4, 0);
#endif
}
#endif

// ---------------------------------------------------------------------------
// fp32 -> bf16 conversion (vector-of-4 per thread)
// ---------------------------------------------------------------------------
__global__ void cvt_f32_bf16(const float* __restrict__ in,
                             __bf16* __restrict__ out, int n) {
    int i = (blockIdx.x * blockDim.x + threadIdx.x) * 4;
    if (i < n) {
        float4 f = *(const float4*)(in + i);
        v4bf o;
        o.x = (__bf16)f.x; o.y = (__bf16)f.y;
        o.z = (__bf16)f.z; o.w = (__bf16)f.w;
        *(v4bf*)(out + i) = o;
    }
}

// ---------------------------------------------------------------------------
// GEMM:  out[m][n] = sum_k A[m][k] * W[n][k] + bias[n]   (torch Linear)
// Block: 256 threads = 8 waves; block tile 128(M) x 128(N);
// wave tile 32(M) x 64(N): B-fragments shared across two A-fragments.
// mode 0: bf16 [B,H,T,Dh] | mode 1: bf16 [B,H,Dh,T] | mode 2: fp32 [M][N]
// ---------------------------------------------------------------------------
__global__ __launch_bounds__(256)
void gemm_bf16_wmma(const __bf16* __restrict__ A,
                    const __bf16* __restrict__ W,
                    const float*  __restrict__ bias,
                    void* __restrict__ outp, int mode) {
    const int lane  = threadIdx.x & 31;
    const int w     = threadIdx.x >> 5;
    const int wm    = w >> 1;          // 0..3
    const int wn    = w & 1;           // 0..1
    const int half  = lane >> 4;
    const int ln16  = lane & 15;
    const int mBase = blockIdx.x * 128 + wm * 32;
    const int nBase = blockIdx.y * 128 + wn * 64;

    v8f acc[2][4] = {};
    const __bf16* arow0 = A + (size_t)(mBase + ln16) * D_MODEL + half * 8;
    const __bf16* arow1 = arow0 + (size_t)16 * D_MODEL;

    for (int kb = 0; kb < D_MODEL; kb += 32) {
        __builtin_prefetch(arow0 + kb + 128, 0, 1);
        v16bf af0 = make_frag(*(const v8bf*)(arow0 + kb),
                              *(const v8bf*)(arow0 + kb + 16));
        v16bf af1 = make_frag(*(const v8bf*)(arow1 + kb),
                              *(const v8bf*)(arow1 + kb + 16));
#pragma unroll
        for (int nt = 0; nt < 4; ++nt) {
            const __bf16* brow = W + (size_t)(nBase + nt * 16 + ln16) * D_MODEL
                                   + kb + half * 8;
            v16bf bf = make_frag(*(const v8bf*)(brow), *(const v8bf*)(brow + 16));
            acc[0][nt] = __builtin_amdgcn_wmma_f32_16x16x32_bf16(
                false, af0, false, bf, (short)0, acc[0][nt], false, false);
            acc[1][nt] = __builtin_amdgcn_wmma_f32_16x16x32_bf16(
                false, af1, false, bf, (short)0, acc[1][nt], false, false);
        }
    }

    float*  fout = (float*)outp;
    __bf16* bout = (__bf16*)outp;
#pragma unroll
    for (int mt = 0; mt < 2; ++mt) {
#pragma unroll
        for (int nt = 0; nt < 4; ++nt) {
            const int n_g = nBase + nt * 16 + ln16;
            const float bv = bias[n_g];
#pragma unroll
            for (int r = 0; r < 8; ++r) {
                const int m_g = mBase + mt * 16 + r + 8 * half;
                const float val = acc[mt][nt][r] + bv;
                if (mode == 2) {
                    fout[(size_t)m_g * D_MODEL + n_g] = val;
                } else {
                    const int b = m_g >> 11, t = m_g & (TSEQ - 1);
                    const int h = n_g >> 6,  d = n_g & (HEAD_DIM - 1);
                    size_t idx;
                    if (mode == 0)
                        idx = (((size_t)b * N_HEADS + h) * TSEQ + t) * HEAD_DIM + d;
                    else
                        idx = (((size_t)b * N_HEADS + h) * HEAD_DIM + d) * TSEQ + t;
                    bout[idx] = (__bf16)val;
                }
            }
        }
    }
}

// ---------------------------------------------------------------------------
// Flash attention, one wave32 per (b,h, 16-query tile).
// K/V tiles staged into LDS by the Tensor Data Mover when available.
// Q,K: [BH][T][64] bf16.  VT: [BH][64][T] bf16.  Y: [B][T][1024] bf16.
// ---------------------------------------------------------------------------
__global__ __launch_bounds__(32)
void flash_attn_wmma(const __bf16* __restrict__ Q,
                     const __bf16* __restrict__ K,
                     const __bf16* __restrict__ VT,
                     __bf16* __restrict__ Y) {
    __shared__ __bf16 kTile[32][64];   // keys x Dh
    __shared__ __bf16 vTile[64][32];   // Dh x keys
    __shared__ __bf16 pTile[16][32];
    const int lane  = threadIdx.x & 31;
    const int half  = lane >> 4;
    const int ln16  = lane & 15;
    const int bh    = blockIdx.y;
    const int qBase = blockIdx.x * 16;

#if HAVE_TDM
    const unsigned int kOff = (unsigned int)(size_t)&kTile[0][0];
    const unsigned int vOff = (unsigned int)(size_t)&vTile[0][0];
#endif

    // Q A-fragments (16 x 64 split into two 16 x 32)
    const __bf16* qrow = Q + ((size_t)bh * TSEQ + qBase + ln16) * HEAD_DIM + half * 8;
    v16bf aq[2];
#pragma unroll
    for (int f = 0; f < 2; ++f)
        aq[f] = make_frag(*(const v8bf*)(qrow + f * 32),
                          *(const v8bf*)(qrow + f * 32 + 16));

    v8f o[4] = {};
    float rowmax[8], rowsum[8];
#pragma unroll
    for (int r = 0; r < 8; ++r) { rowmax[r] = -1e30f; rowsum[r] = 0.0f; }

    const int nChunks = (qBase + 16 + 31) >> 5;
    for (int c = 0; c < nChunks; ++c) {
        const int keyBase = c * 32;
#if HAVE_TDM
        // Previous chunk's ds reads must drain before TDM overwrites the tiles.
        asm volatile("s_wait_dscnt 0" ::: "memory");
        tdm_load_2d(kOff, K + ((size_t)bh * TSEQ + keyBase) * HEAD_DIM,
                    /*tile_x=*/HEAD_DIM, /*tile_y=*/32,
                    /*tensor_x=*/HEAD_DIM, /*tensor_y=*/TSEQ, /*stride=*/HEAD_DIM);
        tdm_load_2d(vOff, VT + (size_t)bh * HEAD_DIM * TSEQ + keyBase,
                    /*tile_x=*/32, /*tile_y=*/HEAD_DIM,
                    /*tensor_x=*/TSEQ, /*tensor_y=*/HEAD_DIM, /*stride=*/TSEQ);
        asm volatile("" ::: "memory");
        __builtin_amdgcn_s_wait_tensorcnt(0);
        asm volatile("" ::: "memory");
#endif
        // ---- S = Q K^T (two 16x16 N-tiles, K=64 contracted as 2x32) ----
        v8f s[2];
#pragma unroll
        for (int nt = 0; nt < 2; ++nt) {
            v8f a = {};
#pragma unroll
            for (int f = 0; f < 2; ++f) {
#if HAVE_TDM
                const __bf16* krow = &kTile[nt * 16 + ln16][f * 32 + half * 8];
#else
                const __bf16* krow = K + ((size_t)bh * TSEQ + keyBase + nt * 16 + ln16)
                                         * HEAD_DIM + f * 32 + half * 8;
#endif
                v16bf bk = make_frag(*(const v8bf*)(krow), *(const v8bf*)(krow + 16));
                a = __builtin_amdgcn_wmma_f32_16x16x32_bf16(
                    false, aq[f], false, bk, (short)0, a, false, false);
            }
            s[nt] = a;
        }
        // ---- scale + causal mask ----
        const float scale = 0.125f;  // 1/sqrt(64)
#pragma unroll
        for (int nt = 0; nt < 2; ++nt) {
            const int key = keyBase + nt * 16 + ln16;
#pragma unroll
            for (int r = 0; r < 8; ++r) {
                const int qi = qBase + r + 8 * half;
                const float v = s[nt][r] * scale;
                s[nt][r] = (key <= qi) ? v : -1e30f;
            }
        }
        // ---- online softmax (row reduce within 16-lane half) ----
        float nm[8], alpha[8];
#pragma unroll
        for (int r = 0; r < 8; ++r) {
            float t = fmaxf(s[0][r], s[1][r]);
#pragma unroll
            for (int m = 1; m < 16; m <<= 1) t = fmaxf(t, __shfl_xor(t, m, 32));
            nm[r] = fmaxf(rowmax[r], t);
            alpha[r] = __expf(rowmax[r] - nm[r]);
            rowmax[r] = nm[r];
        }
#pragma unroll
        for (int r = 0; r < 8; ++r) {
            float p0 = __expf(s[0][r] - nm[r]);
            float p1 = __expf(s[1][r] - nm[r]);
            s[0][r] = p0; s[1][r] = p1;
            float t = p0 + p1;
#pragma unroll
            for (int m = 1; m < 16; m <<= 1) t += __shfl_xor(t, m, 32);
            rowsum[r] = rowsum[r] * alpha[r] + t;
        }
        // ---- C-layout -> A-layout via LDS (single wave; DS in-order) ----
#pragma unroll
        for (int nt = 0; nt < 2; ++nt)
#pragma unroll
            for (int r = 0; r < 8; ++r)
                pTile[r + 8 * half][nt * 16 + ln16] = (__bf16)s[nt][r];
        asm volatile("s_wait_dscnt 0" ::: "memory");
        v16bf pa = make_frag(*(const v8bf*)(&pTile[ln16][half * 8]),
                             *(const v8bf*)(&pTile[ln16][16 + half * 8]));
        // ---- O = O*alpha + P @ V (4 N-tiles over Dh) ----
#pragma unroll
        for (int nt = 0; nt < 4; ++nt) {
#pragma unroll
            for (int r = 0; r < 8; ++r) o[nt][r] *= alpha[r];
#if HAVE_TDM
            const __bf16* vrow = &vTile[nt * 16 + ln16][half * 8];
#else
            const __bf16* vrow = VT + ((size_t)bh * HEAD_DIM + nt * 16 + ln16)
                                      * TSEQ + keyBase + half * 8;
#endif
            v16bf bv = make_frag(*(const v8bf*)(vrow), *(const v8bf*)(vrow + 16));
            o[nt] = __builtin_amdgcn_wmma_f32_16x16x32_bf16(
                false, pa, false, bv, (short)0, o[nt], false, false);
        }
    }
    // ---- epilogue: normalize + store y as bf16 [B,T,C] ----
    const int b = bh >> 4, h = bh & 15;
#pragma unroll
    for (int nt = 0; nt < 4; ++nt) {
        const int d = nt * 16 + ln16;
#pragma unroll
        for (int r = 0; r < 8; ++r) {
            const int t = qBase + r + 8 * half;
            const float val = o[nt][r] / rowsum[r];
            Y[((size_t)b * TSEQ + t) * D_MODEL + h * HEAD_DIM + d] = (__bf16)val;
        }
    }
}

// ---------------------------------------------------------------------------
extern "C" void kernel_launch(void* const* d_in, const int* in_sizes, int n_in,
                              void* d_out, int out_size, void* d_ws, size_t ws_size,
                              hipStream_t stream) {
    (void)in_sizes; (void)n_in; (void)out_size; (void)ws_size;
    const float* x  = (const float*)d_in[0];
    const float* Wq = (const float*)d_in[1];
    const float* bq = (const float*)d_in[2];
    const float* Wk = (const float*)d_in[3];
    const float* bk = (const float*)d_in[4];
    const float* Wv = (const float*)d_in[5];
    const float* bv = (const float*)d_in[6];
    const float* Wp = (const float*)d_in[7];
    const float* bp = (const float*)d_in[8];
    float* out = (float*)d_out;

    char* ws = (char*)d_ws;
    __bf16* xb  = (__bf16*)(ws);                       // 8 MB  [4096][1024]
    __bf16* wqb = (__bf16*)(ws + (size_t)( 8 << 20));  // 2 MB each
    __bf16* wkb = (__bf16*)(ws + (size_t)(10 << 20));
    __bf16* wvb = (__bf16*)(ws + (size_t)(12 << 20));
    __bf16* wpb = (__bf16*)(ws + (size_t)(14 << 20));
    __bf16* qb  = (__bf16*)(ws + (size_t)(16 << 20));  // 8 MB [BH][T][64]
    __bf16* kbf = (__bf16*)(ws + (size_t)(24 << 20));  // 8 MB [BH][T][64]
    __bf16* vtb = (__bf16*)(ws + (size_t)(32 << 20));  // 8 MB [BH][64][T]
    __bf16* yb  = (__bf16*)(ws + (size_t)(40 << 20));  // 8 MB [4096][1024]

    const int nx = MTOT * D_MODEL, nw = D_MODEL * D_MODEL;
    cvt_f32_bf16<<<(nx / 4 + 255) / 256, 256, 0, stream>>>(x,  xb,  nx);
    cvt_f32_bf16<<<(nw / 4 + 255) / 256, 256, 0, stream>>>(Wq, wqb, nw);
    cvt_f32_bf16<<<(nw / 4 + 255) / 256, 256, 0, stream>>>(Wk, wkb, nw);
    cvt_f32_bf16<<<(nw / 4 + 255) / 256, 256, 0, stream>>>(Wv, wvb, nw);
    cvt_f32_bf16<<<(nw / 4 + 255) / 256, 256, 0, stream>>>(Wp, wpb, nw);

    dim3 g(MTOT / 128, D_MODEL / 128);
    gemm_bf16_wmma<<<g, 256, 0, stream>>>(xb, wqb, bq, (void*)qb,  0);
    gemm_bf16_wmma<<<g, 256, 0, stream>>>(xb, wkb, bk, (void*)kbf, 0);
    gemm_bf16_wmma<<<g, 256, 0, stream>>>(xb, wvb, bv, (void*)vtb, 1);

    flash_attn_wmma<<<dim3(TSEQ / 16, BATCH * N_HEADS), 32, 0, stream>>>(
        qb, kbf, vtb, yb);

    gemm_bf16_wmma<<<g, 256, 0, stream>>>(yb, wpb, bp, (void*)out, 2);
}
